// VectorQuantizerEMA_2130303779122
// MI455X (gfx1250) — matrix operand order repaired and checked by reading, not compiled, not run on hardware
//
#include <hip/hip_runtime.h>
#include <hip/hip_bf16.h>

#define NUM_CODES 1024
#define CODE_DIM  256
#define COMMIT_W  0.25f
#define EPS_      1e-5f
#define ENT_LO    0.5f
#define ENT_HI    0.9f
#define ENT_W     0.1f
#define VAR_FLOOR 0.05f
#define VAR_W     1e-3f
#define DECORR_W  1e-3f

// padded LDS tile: 16 codes x 264 halves (256 data + 8 pad) -> conflict-free b128 reads
#define TROW 264

typedef __attribute__((ext_vector_type(16))) _Float16 v16h;
typedef __attribute__((ext_vector_type(8)))  _Float16 v8h;
typedef __attribute__((ext_vector_type(8)))  float    v8f;

union V16H { v16h v; v8h h2[2]; _Float16 h[16]; };

__device__ __forceinline__ void wait_asynccnt0() {
#if __has_builtin(__builtin_amdgcn_s_wait_asynccnt)
    __builtin_amdgcn_s_wait_asynccnt(0);
#else
    asm volatile("s_wait_asynccnt 0" ::: "memory");
#endif
}

__device__ __forceinline__ void async_lds_b128(unsigned lds_byte_addr,
                                               unsigned long long gaddr) {
    // VFLAT async: VDST = LDS byte address VGPR, VADDR = 64-bit global address
    asm volatile("global_load_async_to_lds_b128 %0, %1, off"
                 :: "v"(lds_byte_addr), "v"(gaddr)
                 : "memory");
}

// ---------------------------------------------------------------- init
__global__ void init_accums(double* acc) {
    if (threadIdx.x == 0) { acc[0] = 0.0; acc[1] = 0.0; }
}

// ------------------------------------------- per-dim mean + variance penalty
__global__ void colstats_kernel(const float* __restrict__ W,
                                float* __restrict__ colmean,
                                float* __restrict__ varsum) {
    const int d = threadIdx.x;                 // 256 threads, one block
    float s = 0.f, s2 = 0.f;
    for (int k = 0; k < NUM_CODES; ++k) {
        float w = W[(size_t)k * CODE_DIM + d];
        s += w; s2 += w * w;
    }
    const float m   = s  * (1.f / NUM_CODES);
    const float var = s2 * (1.f / NUM_CODES) - m * m;
    colmean[d] = m;
    __shared__ float red[CODE_DIM];
    red[d] = fmaxf(VAR_FLOOR - var, 0.f);
    __syncthreads();
    for (int off = CODE_DIM / 2; off > 0; off >>= 1) {
        if (d < off) red[d] += red[d + off];
        __syncthreads();
    }
    if (d == 0) *varsum = red[0];
}

// ---------------- f16 codebook copy, centered+transposed copy, code norms
__global__ void prep_kernel(const float* __restrict__ W,
                            const float* __restrict__ colmean,
                            _Float16* __restrict__ wh,
                            _Float16* __restrict__ wcT,
                            float* __restrict__ wnorm) {
    const int k = blockIdx.x;                  // code
    const int d = threadIdx.x;                 // dim
    const float w = W[(size_t)k * CODE_DIM + d];
    wh [(size_t)k * CODE_DIM + d] = (_Float16)w;
    wcT[(size_t)d * NUM_CODES + k] = (_Float16)(w - colmean[d]);
    __shared__ float red[CODE_DIM];
    red[d] = w * w;
    __syncthreads();
    for (int off = CODE_DIM / 2; off > 0; off >>= 1) {
        if (d < off) red[d] += red[d + off];
        __syncthreads();
    }
    if (d == 0) wnorm[k] = red[0];
}

// ------------------------------------- main: WMMA scores + argmin + gather
__global__ __launch_bounds__(256)
void vq_main_kernel(const float* __restrict__ X, const float* __restrict__ W,
                    const _Float16* __restrict__ wh, const float* __restrict__ wnorm,
                    float* __restrict__ quant, float* __restrict__ idx_out,
                    double* __restrict__ accCommit) {
    const int lane = threadIdx.x & 31;
    const int wave = threadIdx.x >> 5;
    const int half = lane >> 4;
    const int l16  = lane & 15;
    const int rowbase = (blockIdx.x * 8 + wave) * 16;   // 16 rows per wave

    // double-buffered codebook tile (async-staged) + wnorm table in LDS
    __shared__ _Float16 atile[2][16 * TROW];
    __shared__ float    wn_lds[NUM_CODES];

    // stage wnorm once (4KB, L2-hot broadcast -> LDS)
    #pragma unroll
    for (int j = 0; j < 4; ++j) {
        int i = threadIdx.x + j * 256;
        wn_lds[i] = wnorm[i];
    }

    // cooperative async stage of one 8KB codebook tile into buffer (t&1):
    // 512 chunks of 16B; 256 threads x 2 chunks, row-padded to TROW halves
    auto issue_tile = [&](int t) {
        const unsigned long long gbase =
            (unsigned long long)(uintptr_t)wh + (unsigned long long)t * 8192ull;
        #pragma unroll
        for (int q = 0; q < 2; ++q) {
            const int g   = threadIdx.x + q * 256;   // chunk id 0..511
            const int row = g >> 5;                  // 32 chunks per 512B row
            const int col = g & 31;
            unsigned lds = (unsigned)(uintptr_t)(&atile[t & 1][row * TROW + col * 8]);
            async_lds_b128(lds, gbase + (unsigned long long)g * 16ull);
        }
    };

    // B = input rows (K x 16 rows), register-resident, f32->f16 once.
    // B layout: lane (n, half) holds column n = row rowbase+n, K = 16*half + e.
    v16h breg[8];
    {
        const float* src = X + (size_t)(rowbase + l16) * CODE_DIM;
        #pragma unroll
        for (int c = 0; c < 8; ++c) {
            V16H b;
            const float* p = src + c * 32 + 16 * half;
            #pragma unroll
            for (int e = 0; e < 16; ++e) b.h[e] = (_Float16)p[e];
            breg[c] = b.v;
        }
    }

    issue_tile(0);

    float best = 3.4e38f;
    int   bidx = 0;
    for (int t = 0; t < NUM_CODES / 16; ++t) {
        wait_asynccnt0();       // this wave's staged chunks landed in LDS
        __syncthreads();        // everyone's chunks landed; prev buffer free
        if (t + 1 < NUM_CODES / 16) issue_tile(t + 1);

        v8f acc = {};
        // A = codebook tile from LDS: row m = code t*16+l16 (contiguous K = 16*half + e)
        const _Float16* arow = &atile[t & 1][l16 * TROW];
        #pragma unroll
        for (int c = 0; c < 8; ++c) {
            V16H a;
            a.h2[0] = *(const v8h*)(arow + c * 32 + 16 * half);
            a.h2[1] = *(const v8h*)(arow + c * 32 + 16 * half + 8);
            acc = __builtin_amdgcn_wmma_f32_16x16x32_f16(
                      false, a.v, false, breg[c], (short)0, acc, false, false);
        }
        // score[k] = ||w_k||^2 - 2 x.w_k   (||x||^2 irrelevant for argmin)
        const float* wn = &wn_lds[t * 16 + 8 * half];
        #pragma unroll
        for (int r = 0; r < 8; ++r) {
            float s = wn[r] - 2.0f * acc[r];
            int code = t * 16 + 8 * half + r;
            if (s < best) { best = s; bidx = code; }
        }
    }
    // merge the two half-lane partial argmins for row rowbase+l16
    {
        float ob = __shfl_xor(best, 16);
        int   oi = __shfl_xor(bidx, 16);
        if (ob < best || (ob == best && oi < bidx)) { best = ob; bidx = oi; }
    }
    if (lane < 16) idx_out[rowbase + l16] = (float)bidx;

    // gather quant = weight[idx] (f32) + commitment partial
    float csum = 0.f;
    const int d0 = lane * 8;
    for (int r = 0; r < 16; ++r) {
        const int code = __shfl(bidx, r);
        const float* qsrc = W + (size_t)code * CODE_DIM + d0;
        const float* xsrc = X + (size_t)(rowbase + r) * CODE_DIM + d0;
        float*       qdst = quant + (size_t)(rowbase + r) * CODE_DIM + d0;
        float4 q0 = *(const float4*)(qsrc);
        float4 q1 = *(const float4*)(qsrc + 4);
        float4 x0 = *(const float4*)(xsrc);
        float4 x1 = *(const float4*)(xsrc + 4);
        *(float4*)(qdst)     = q0;
        *(float4*)(qdst + 4) = q1;
        float dx;
        dx = q0.x - x0.x; csum += dx * dx;  dx = q0.y - x0.y; csum += dx * dx;
        dx = q0.z - x0.z; csum += dx * dx;  dx = q0.w - x0.w; csum += dx * dx;
        dx = q1.x - x1.x; csum += dx * dx;  dx = q1.y - x1.y; csum += dx * dx;
        dx = q1.z - x1.z; csum += dx * dx;  dx = q1.w - x1.w; csum += dx * dx;
    }
    for (int off = 16; off >= 1; off >>= 1) csum += __shfl_xor(csum, off);
    if (lane == 0) atomicAdd(accCommit, (double)csum);
}

// ------------------------------------- decorrelation: cov = wc^T wc via WMMA
__global__ __launch_bounds__(32)
void decorr_kernel(const _Float16* __restrict__ wcT, double* __restrict__ accDecorr) {
    const int lane = threadIdx.x & 31;
    const int half = lane >> 4;
    const int l16  = lane & 15;
    const int i0 = (blockIdx.x >> 4) * 16;   // output row-dim tile
    const int j0 = (blockIdx.x & 15) * 16;   // output col-dim tile
    v8f acc = {};
    for (int kc = 0; kc < NUM_CODES / 32; ++kc) {
        const int kb = kc * 32;
        V16H a;   // A row m = dim i0+l16 of wcT (contiguous in k)
        const _Float16* ap = wcT + (size_t)(i0 + l16) * NUM_CODES + kb;
        a.h2[0] = *(const v8h*)(ap + 16 * half);
        a.h2[1] = *(const v8h*)(ap + 16 * half + 8);
        v16h b = *(const v16h*)(wcT + (size_t)(j0 + l16) * NUM_CODES + kb + 16 * half);
        acc = __builtin_amdgcn_wmma_f32_16x16x32_f16(
                  false, a.v, false, b, (short)0, acc, false, false);
    }
    float local = 0.f;
    #pragma unroll
    for (int r = 0; r < 8; ++r) {
        const int i = i0 + 8 * half + r;
        const int j = j0 + l16;
        float cov = acc[r] * (1.0f / NUM_CODES);
        if (i != j) local += cov * cov;
    }
    for (int off = 16; off >= 1; off >>= 1) local += __shfl_xor(local, off);
    if (lane == 0) atomicAdd(accDecorr, (double)local);
}

// ---------------------------------------------------------------- finalize
__global__ void finalize_kernel(const float* __restrict__ usage,
                                const double* __restrict__ acc,
                                const float* __restrict__ varsum,
                                float* __restrict__ out_scalars,
                                float inv_n_elems) {
    __shared__ float red[NUM_CODES];
    const int t = threadIdx.x;
    const float u = usage[t] + EPS_;
    red[t] = u;
    __syncthreads();
    for (int off = NUM_CODES / 2; off > 0; off >>= 1) {
        if (t < off) red[t] += red[t + off];
        __syncthreads();
    }
    const float S = red[0];
    __syncthreads();
    const float p = u / fmaxf(S, EPS_ * NUM_CODES);
    red[t] = -p * logf(p + EPS_);
    __syncthreads();
    for (int off = NUM_CODES / 2; off > 0; off >>= 1) {
        if (t < off) red[t] += red[t + off];
        __syncthreads();
    }
    if (t == 0) {
        const float H = red[0] / logf((float)NUM_CODES);
        float gap = 0.f;
        if (H < ENT_LO) gap = ENT_LO - H;
        else if (H > ENT_HI) gap = H - ENT_HI;
        const float ent_loss = ENT_W * gap * gap;
        const float commit   = COMMIT_W * (float)(acc[0] * (double)inv_n_elems);
        const float var_loss = VAR_W * (*varsum) * (1.f / CODE_DIM);
        const float dec_loss = DECORR_W * (float)acc[1] * (1.f / ((float)CODE_DIM * CODE_DIM));
        const float total = commit + ent_loss + var_loss + dec_loss;
        out_scalars[0] = total;
        out_scalars[1] = commit;
        out_scalars[2] = ent_loss;
        out_scalars[3] = var_loss;
        out_scalars[4] = dec_loss;
        out_scalars[5] = H;
    }
}

extern "C" void kernel_launch(void* const* d_in, const int* in_sizes, int n_in,
                              void* d_out, int out_size, void* d_ws, size_t ws_size,
                              hipStream_t stream) {
    const float* X     = (const float*)d_in[0];   // [N, 256]
    const float* W     = (const float*)d_in[1];   // [1024, 256]
    const float* usage = (const float*)d_in[2];   // [1024]
    const int N = in_sizes[0] / CODE_DIM;         // 32768

    char* ws = (char*)d_ws;
    _Float16* wh    = (_Float16*)(ws);                 // 512 KB f16 codebook
    _Float16* wcT   = (_Float16*)(ws + 524288);        // 512 KB centered^T [256][1024]
    float*    wnorm = (float*)(ws + 1048576);          // 4 KB
    float*    cmean = (float*)(ws + 1052672);          // 1 KB
    double*   acc   = (double*)(ws + 1053696);         // [commit, decorr]
    float*    vsum  = (float*)(ws + 1053712);

    float* out     = (float*)d_out;
    float* quant   = out;                              // N*256
    float* idxs    = out + (size_t)N * CODE_DIM;       // N
    float* scalars = idxs + N;                         // 6

    init_accums    <<<1, 32, 0, stream>>>(acc);
    colstats_kernel<<<1, CODE_DIM, 0, stream>>>(W, cmean, vsum);
    prep_kernel    <<<NUM_CODES, CODE_DIM, 0, stream>>>(W, cmean, wh, wcT, wnorm);
    vq_main_kernel <<<N / 128, 256, 0, stream>>>(X, W, wh, wnorm, quant, idxs, &acc[0]);
    decorr_kernel  <<<256, 32, 0, stream>>>(wcT, &acc[1]);
    finalize_kernel<<<1, NUM_CODES, 0, stream>>>(usage, acc, vsum, scalars,
                                                 1.0f / ((float)N * (float)CODE_DIM));
}
